// TemporalBlock_63599875719567
// MI455X (gfx1250) — compile-verified, matrix operand order checked by reading, not run on hardware
//
#include <hip/hip_runtime.h>
#include <hip/hip_bf16.h>

// ---------------------------------------------------------------------------
// Mamba-style selective SSM (last-step output) for MI455X / gfx1250.
//
// One workgroup (256 threads = 8 wave32) per sequence (5200 sequences):
//   1) u(96x128), W_in[0:256](256x128) -> LDS f16
//   2) z_last = u[95] @ W_in[256:512]^T                (VALU GEMV)
//   3) xp(96x256) = u @ W_in[0:256]^T                  (WMMA, B-frags in regs)
//   4) causal depthwise conv4 + SiLU in place          (VALU)
//   5) xdbl(96x48) = xc @ W_xproj^T (40 padded to 48)  (WMMA, A-frag reuse x3)
//   6) 96-step selective scan, h in regs               (v_exp_f32 via exp2)
//   7) y = (h @ Cc_last + D*xc_last) * silu(z_last); out = W_out @ y
// ---------------------------------------------------------------------------

typedef __attribute__((ext_vector_type(16))) _Float16 v16h;
typedef __attribute__((ext_vector_type(8)))  float    v8f;

#define T_LEN   96
#define SEQS    5200
#define SA_STR  136   // u rows: 128 + 8 pad halves (kills bank conflicts on b128)
#define SB_STR  136   // W_in rows
#define SC_STR  264   // xc rows: 256 + 8 pad halves
#define SXP_STR 264   // W_xproj rows (reuses sA)
#define XD_STR  52    // xdbl f32 row stride (48 cols + pad)
#define LOG2E   1.4426950408889634f

#define WMMA_F16(A, B, C) \
  __builtin_amdgcn_wmma_f32_16x16x32_f16(false, (A), false, (B), (short)0, (C), false, false)

// Load one 16x32 f16 WMMA operand fragment from LDS (row-major, padded).
// CDNA5 ISA 7.12.2: lanes 0-15 hold M=lane, K in [kb,kb+8)+[kb+16,kb+24),
// kb = kt*32; lanes 16-31 same rows, kb += 8.  Two ds_load_b128 per fragment.
__device__ __forceinline__ v16h load_frag16(const _Float16* base, int row0,
                                            int stride, int kt, int lane) {
  const int m  = lane & 15;
  const int kb = kt * 32 + ((lane & 16) ? 8 : 0);
  const _Float16* p = base + (row0 + m) * stride + kb;
  v16h f;
  ((uint4*)&f)[0] = *(const uint4*)(p);        // K = kb .. kb+7
  ((uint4*)&f)[1] = *(const uint4*)(p + 16);   // K = kb+16 .. kb+23
  return f;
}

__global__ __launch_bounds__(256)
void mamba_last_kernel(const float* __restrict__ x,      const float* __restrict__ W_in,
                       const float* __restrict__ conv_w, const float* __restrict__ conv_b,
                       const float* __restrict__ W_xproj,const float* __restrict__ W_dt,
                       const float* __restrict__ b_dt,   const float* __restrict__ A_log,
                       const float* __restrict__ Dp,     const float* __restrict__ W_out,
                       float* __restrict__ out)
{
  __shared__ __align__(16) _Float16 sA[T_LEN * SA_STR];   // u f16 -> later W_xproj f16 (48x264)
  __shared__ __align__(16) _Float16 sB[256   * SB_STR];   // W_in f16 -> later xdbl f32 (96x52)
  __shared__ __align__(16) _Float16 sC[T_LEN * SC_STR];   // xp f16 -> xc f16 (in-place conv)
  __shared__ float zbuf[256];
  __shared__ float ybuf[256];
  __shared__ float red[256];

  const int tid  = threadIdx.x;
  const int lane = tid & 31;
  const int wave = tid >> 5;
  const int seq  = blockIdx.x;
  const float* xs = x + (size_t)seq * (T_LEN * 128);

  // ---- stage u and W_in[0:256] into LDS as f16 -----------------------------
  for (int i = tid; i < T_LEN * 128; i += 256) {
    int t = i >> 7, k = i & 127;
    sA[t * SA_STR + k] = (_Float16)xs[i];
  }
  for (int i = tid; i < 256 * 128; i += 256) {
    int e = i >> 7, k = i & 127;
    sB[e * SB_STR + k] = (_Float16)W_in[i];
  }
  __syncthreads();

  // ---- z_last[e] = dot(u[95,:], W_in[256+e,:])  (only t=95 of z is used) ---
  {
    const float* wr = W_in + (size_t)(256 + tid) * 128;
    float acc = 0.f;
    #pragma unroll 4
    for (int k = 0; k < 128; ++k)
      acc = fmaf((float)sA[95 * SA_STR + k], wr[k], acc);
    zbuf[tid] = acc;
  }

  // ---- GEMM1: xp(96x256) = u @ W_in[0:256]^T -------------------------------
  // Each wave owns 2 N-tiles; B fragments (2 nt x 4 kt) stay in registers,
  // A fragments loaded once per (mt,kt) and feed two independent WMMAs.
  {
    const int nt0 = wave * 2, nt1 = nt0 + 1;
    v16h bf0[4], bf1[4];
    #pragma unroll
    for (int kt = 0; kt < 4; ++kt) {
      bf0[kt] = load_frag16(sB, nt0 * 16, SB_STR, kt, lane);
      bf1[kt] = load_frag16(sB, nt1 * 16, SB_STR, kt, lane);
    }
    const int lm = (lane >> 4) * 8, ln = lane & 15;
    for (int mt = 0; mt < 6; ++mt) {
      v16h af[4];
      #pragma unroll
      for (int kt = 0; kt < 4; ++kt)
        af[kt] = load_frag16(sA, mt * 16, SA_STR, kt, lane);
      v8f c0 = {}, c1 = {};
      #pragma unroll
      for (int kt = 0; kt < 4; ++kt) {
        c0 = WMMA_F16(af[kt], bf0[kt], c0);
        c1 = WMMA_F16(af[kt], bf1[kt], c1);
      }
      #pragma unroll
      for (int r = 0; r < 8; ++r) {
        sC[(mt * 16 + lm + r) * SC_STR + nt0 * 16 + ln] = (_Float16)c0[r];
        sC[(mt * 16 + lm + r) * SC_STR + nt1 * 16 + ln] = (_Float16)c1[r];
      }
    }
  }
  __syncthreads();

  // ---- causal depthwise conv (k=4) + SiLU, in place on column d=tid --------
  {
    const float w0 = conv_w[tid * 4 + 0], w1 = conv_w[tid * 4 + 1];
    const float w2 = conv_w[tid * 4 + 2], w3 = conv_w[tid * 4 + 3];
    const float cb = conv_b[tid];
    float h0 = 0.f, h1 = 0.f, h2 = 0.f;
    for (int t = 0; t < T_LEN; ++t) {
      float xv = (float)sC[t * SC_STR + tid];
      float a  = fmaf(w0, h0, fmaf(w1, h1, fmaf(w2, h2, fmaf(w3, xv, cb))));
      float sc = a / (1.f + __expf(-a));                 // SiLU
      sC[t * SC_STR + tid] = (_Float16)sc;
      h0 = h1; h1 = h2; h2 = xv;
    }
  }
  // ---- stage W_xproj (40x256, zero-padded to 48 rows) into sA --------------
  for (int i = tid; i < 48 * 256; i += 256) {
    int n = i >> 8, k = i & 255;
    float v = (n < 40) ? W_xproj[n * 256 + k] : 0.f;
    sA[n * SXP_STR + k] = (_Float16)v;
  }
  __syncthreads();

  // ---- GEMM2: xdbl(96x48) = xc @ W_xproj^T ---------------------------------
  // Waves 0..5 each own one M-tile (wave-uniform branch -> EXEC stays all-1s);
  // per kt, one A fragment feeds three independent WMMAs (3 N-tiles).
  float* xdbl = (float*)sB;                   // reuse W_in region
  if (wave < 6) {
    const int mt = wave;
    v8f c0 = {}, c1 = {}, c2 = {};
    #pragma unroll
    for (int kt = 0; kt < 8; ++kt) {
      v16h a  = load_frag16(sC, mt * 16, SC_STR,  kt, lane);
      v16h b0 = load_frag16(sA, 0,       SXP_STR, kt, lane);
      v16h b1 = load_frag16(sA, 16,      SXP_STR, kt, lane);
      v16h b2 = load_frag16(sA, 32,      SXP_STR, kt, lane);
      c0 = WMMA_F16(a, b0, c0);
      c1 = WMMA_F16(a, b1, c1);
      c2 = WMMA_F16(a, b2, c2);
    }
    const int lm = (lane >> 4) * 8, ln = lane & 15;
    #pragma unroll
    for (int r = 0; r < 8; ++r) {
      xdbl[(mt * 16 + lm + r) * XD_STR +      ln] = c0[r];
      xdbl[(mt * 16 + lm + r) * XD_STR + 16 + ln] = c1[r];
      xdbl[(mt * 16 + lm + r) * XD_STR + 32 + ln] = c2[r];
    }
  }
  __syncthreads();

  // ---- selective scan: thread d owns h[d, 0..15] ---------------------------
  {
    const int d = tid;
    float wdt[8];
    #pragma unroll
    for (int r = 0; r < 8; ++r) wdt[r] = W_dt[d * 8 + r];
    const float bdt = b_dt[d];
    // A pre-scaled by log2(e): dA = exp(dt*A) = exp2(dt * Av2)  (bare v_exp_f32)
    float Av2[16];
    #pragma unroll
    for (int s = 0; s < 16; ++s) Av2[s] = -LOG2E * __expf(A_log[d * 16 + s]);
    float h[16];
    #pragma unroll
    for (int s = 0; s < 16; ++s) h[s] = 0.f;

    for (int t = 0; t < T_LEN; ++t) {
      const float* row = xdbl + t * XD_STR;
      // dt = softplus(dt_r @ W_dt[d] + b_dt[d])   (dt_r = xdbl cols 0..7)
      float4 r0 = *(const float4*)(row);
      float4 r1 = *(const float4*)(row + 4);
      float acc = bdt;
      acc = fmaf(r0.x, wdt[0], acc); acc = fmaf(r0.y, wdt[1], acc);
      acc = fmaf(r0.z, wdt[2], acc); acc = fmaf(r0.w, wdt[3], acc);
      acc = fmaf(r1.x, wdt[4], acc); acc = fmaf(r1.y, wdt[5], acc);
      acc = fmaf(r1.z, wdt[6], acc); acc = fmaf(r1.w, wdt[7], acc);
      float dtv = fmaxf(acc, 0.f) + __logf(1.f + __expf(-fabsf(acc)));
      float xcv = (float)sC[t * SC_STR + d];
      float dux = dtv * xcv;
      // Bc = xdbl cols 8..23 (broadcast float4 LDS loads)
      float4 b0 = *(const float4*)(row + 8);
      float4 b1 = *(const float4*)(row + 12);
      float4 b2 = *(const float4*)(row + 16);
      float4 b3 = *(const float4*)(row + 20);
      float Bv[16] = { b0.x, b0.y, b0.z, b0.w, b1.x, b1.y, b1.z, b1.w,
                       b2.x, b2.y, b2.z, b2.w, b3.x, b3.y, b3.z, b3.w };
      #pragma unroll
      for (int s = 0; s < 16; ++s)
        h[s] = fmaf(h[s], __builtin_amdgcn_exp2f(dtv * Av2[s]), dux * Bv[s]);
    }

    // y[d] = h . Cc_last + D[d]*xc_last[d];  y *= silu(z_last[d])
    float y = Dp[d] * (float)sC[95 * SC_STR + d];
    const float* c95 = xdbl + 95 * XD_STR + 24;    // Cc = cols 24..39
    #pragma unroll
    for (int s = 0; s < 16; ++s) y = fmaf(h[s], c95[s], y);
    const float z = zbuf[d];
    y *= z / (1.f + __expf(-z));
    ybuf[d] = y;
  }
  __syncthreads();

  // ---- out[f] = sum_d W_out[f,d] * y[d]  (split-K over 2 thread groups) ----
  {
    const int f = tid & 127, part = tid >> 7;
    const float* wr = W_out + (size_t)f * 256 + part * 128;
    const float* yv = ybuf + part * 128;
    float acc = 0.f;
    #pragma unroll 4
    for (int d = 0; d < 128; ++d) acc = fmaf(wr[d], yv[d], acc);
    red[tid] = acc;
  }
  __syncthreads();
  if (tid < 128)
    out[(size_t)seq * 128 + tid] = red[tid] + red[128 + tid];
}

extern "C" void kernel_launch(void* const* d_in, const int* in_sizes, int n_in,
                              void* d_out, int out_size, void* d_ws, size_t ws_size,
                              hipStream_t stream) {
  (void)in_sizes; (void)n_in; (void)out_size; (void)d_ws; (void)ws_size;
  const float* x       = (const float*)d_in[0];
  const float* W_in    = (const float*)d_in[1];
  const float* conv_w  = (const float*)d_in[2];
  const float* conv_b  = (const float*)d_in[3];
  const float* W_xproj = (const float*)d_in[4];
  const float* W_dt    = (const float*)d_in[5];
  const float* b_dt    = (const float*)d_in[6];
  const float* A_log   = (const float*)d_in[7];
  const float* Dvec    = (const float*)d_in[8];
  const float* W_out   = (const float*)d_in[9];

  mamba_last_kernel<<<SEQS, 256, 0, stream>>>(x, W_in, conv_w, conv_b, W_xproj,
                                              W_dt, b_dt, A_log, Dvec, W_out,
                                              (float*)d_out);
}
